// PointNet2SemSeg_31576599560324
// MI455X (gfx1250) — matrix operand order repaired and checked by reading, not compiled, not run on hardware
//
#include <hip/hip_runtime.h>

typedef __attribute__((ext_vector_type(16))) _Float16 v16h;
typedef __attribute__((ext_vector_type(8)))  _Float16 v8h;
typedef __attribute__((ext_vector_type(8)))  float    v8f;
typedef _Float16 f16;

#define NUM_CLASSES 9
#define BATCH 4
#define NPTS 8192

// ---------------------------------------------------------------------------
// Activation storage: WMMA-B fragment-native tiling.
// Element (c, x) of the [Cpad][X] channel-major tensor lives at
//   tile  = (c>>5) * (X>>4) + (x>>4)
//   off   = tile*512 + (x&15)*32 + (c&31)        (halves)
// so a wave's 32x16 B fragment is two aligned 16B loads per lane, and the
// GEMM epilogue writes 8 consecutive halves per lane as one 16B store.
// ---------------------------------------------------------------------------
static __device__ __forceinline__ size_t tiled_off(int c, long long x, long long X) {
  return ((size_t)(c >> 5) * (size_t)(X >> 4) + (size_t)(x >> 4)) * 512
       + (size_t)(x & 15) * 32 + (size_t)(c & 31);
}

// A: 16x32 f16 row-major weights; row M = lane%16; lanes 0-15 hold K in
// [0..7]U[16..23], lanes 16-31 hold K in [8..15]U[24..31]. Two 16B loads.
static __device__ __forceinline__ v16h load_a_frag(const f16* __restrict__ W,
                                                   int Cpad, int o0, int c0, int lane) {
  int hi = lane >> 4;
  int m  = lane & 15;
  const f16* row = W + (size_t)(o0 + m) * Cpad + c0 + hi * 8;   // 16B aligned
  v8h lo = *(const v8h*)(row);
  v8h hv = *(const v8h*)(row + 16);
  return __builtin_shufflevector(lo, hv, 0, 1, 2, 3, 4, 5, 6, 7,
                                 8, 9, 10, 11, 12, 13, 14, 15);
}

// B: 32x16 f16 from tiled activations; col N = lane%16; lanes 0-15 hold
// K in [0..7]U[16..23], lanes 16-31 hold K in [8..15]U[24..31]. Two aligned 16B loads per lane.
static __device__ __forceinline__ v16h load_b_frag(const f16* __restrict__ In,
                                                   long long X, int c0, long long x0, int lane) {
  const f16* tile = In + ((size_t)(c0 >> 5) * (size_t)(X >> 4) + (size_t)(x0 >> 4)) * 512;
  const f16* p = tile + (size_t)(lane & 15) * 32 + (size_t)(lane >> 4) * 8;
  v8h lo = *(const v8h*)(p);
  v8h hv = *(const v8h*)(p + 16);
  return __builtin_shufflevector(lo, hv, 0, 1, 2, 3, 4, 5, 6, 7,
                                 8, 9, 10, 11, 12, 13, 14, 15);
}

// ---------------------------------------------------------------------------
// Fused GEMM + bias + affine-BN + ReLU epilogue.
// D[Opad x X] = W[Opad x Cpad] * In_tiled[Cpad x X].
// Wave tile: (16*T) rows x 16 cols; waves of a block take adjacent column
// slices (A redundant -> L2-resident weights; B streamed once).
// gain==nullptr -> bias only, no ReLU (head2). Pad rows written as 0.
// ---------------------------------------------------------------------------
template <int T>
__global__ void __launch_bounds__(128)
gemm_bn_relu_kernel(const f16* __restrict__ Wp, const f16* __restrict__ In,
                    const float* __restrict__ bias, const float* __restrict__ gain,
                    const float* __restrict__ beta,
                    f16* __restrict__ OutH, float* __restrict__ OutF,
                    int Opad, int Cpad, long long X, int Oreal) {
  int lane = threadIdx.x;                                       // 0..31 (wave32)
  int o0 = blockIdx.y * (16 * T);
  long long x0 = ((long long)blockIdx.x * 4 + threadIdx.y) * 16;
  if (o0 >= Opad) return;                                       // wave-uniform

  v8f zero = {};
  v8f acc[T];
#pragma unroll
  for (int t = 0; t < T; ++t) acc[t] = zero;

#pragma unroll 2
  for (int c0 = 0; c0 < Cpad; c0 += 32) {
    if (c0 + 32 < Cpad) {   // prefetch next B k-tile
      const f16* pf = In + ((size_t)((c0 + 32) >> 5) * (size_t)(X >> 4)
                            + (size_t)(x0 >> 4)) * 512 + (size_t)(lane & 15) * 32;
      __builtin_prefetch((const void*)pf, 0, 3);
    }
    v16h b = load_b_frag(In, X, c0, x0, lane);
#pragma unroll
    for (int t = 0; t < T; ++t) {
      v16h a = load_a_frag(Wp, Cpad, o0 + 16 * t, c0, lane);
      acc[t] = __builtin_amdgcn_wmma_f32_16x16x32_f16(false, a, false, b, (short)0,
                                                      acc[t], false, false);
    }
  }

  int hi = lane >> 4;
  long long x = x0 + (lane & 15);
#pragma unroll
  for (int t = 0; t < T; ++t) {
    int ob = o0 + 16 * t + hi * 8;          // 8-aligned, no 32-boundary crossing
    v8h hv;
#pragma unroll
    for (int r = 0; r < 8; ++r) {
      int o = ob + r;
      float v = acc[t][r];
      if (o < Oreal) {
        v += bias[o];
        if (gain) { v = v * gain[o] + beta[o]; v = fmaxf(v, 0.0f); }
      } else {
        v = 0.0f;
      }
      hv[r] = (f16)v;
      if (OutF) OutF[(size_t)o * X + x] = v;
    }
    if (OutH) *(v8h*)(OutH + tiled_off(ob, x, X)) = hv;   // one 16B store
  }
}

// Convert f32 weights [Oreal x Creal] -> zero-padded f16 [Opad x Cpad]
__global__ void convert_weight_kernel(const float* __restrict__ W, f16* __restrict__ Wp,
                                      int Oreal, int Creal, int Opad, int Cpad) {
  int i = blockIdx.x * blockDim.x + threadIdx.x;
  if (i >= Opad * Cpad) return;
  int o = i / Cpad, c = i % Cpad;
  float v = (o < Oreal && c < Creal) ? W[(size_t)o * Creal + c] : 0.0f;
  Wp[i] = (f16)v;
}

// x [B][3][N] -> xyz [B][N][3]
__global__ void transpose_xyz_kernel(const float* __restrict__ x, float* __restrict__ xyz) {
  int t = blockIdx.x * blockDim.x + threadIdx.x;
  if (t >= BATCH * NPTS) return;
  int b = t / NPTS, p = t % NPTS;
  const float* src = x + (size_t)b * 3 * NPTS + p;
  float* dst = xyz + (size_t)t * 3;
  dst[0] = src[0 * NPTS];
  dst[1] = src[1 * NPTS];
  dst[2] = src[2 * NPTS];
}

// Farthest point sampling: one block per batch; dist[] in LDS; wave32-shuffle
// argmax (5 xor steps, no barrier) + one cross-wave LDS step per iteration.
__global__ void fps_kernel(const float* __restrict__ xyz, int n, int npoint,
                           int* __restrict__ fidx) {
  extern __shared__ unsigned char fps_smem[];
  float* dist = (float*)fps_smem;            // n floats
  float* wmax = dist + n;                    // 32 floats
  int*   warg = (int*)(wmax + 32);           // 32 ints
  __shared__ int s_far;
  int b = blockIdx.x;
  int tid = threadIdx.x;
  int nthreads = blockDim.x;
  int lane = tid & 31, wid = tid >> 5;
  int nw = nthreads >> 5;
  const float* P = xyz + (size_t)b * n * 3;
  for (int i = tid; i < n; i += nthreads) dist[i] = 1e10f;
  if (tid == 0) s_far = 0;
  __syncthreads();
  for (int it = 0; it < npoint; ++it) {
    int far = s_far;
    if (tid == 0) fidx[(size_t)b * npoint + it] = far;
    float cx = P[(size_t)far * 3], cy = P[(size_t)far * 3 + 1], cz = P[(size_t)far * 3 + 2];
    float best = -1.0f; int bi = n - 1;
    for (int i = tid; i < n; i += nthreads) {
      float dx = P[(size_t)i * 3] - cx;
      float dy = P[(size_t)i * 3 + 1] - cy;
      float dz = P[(size_t)i * 3 + 2] - cz;
      float d = dx * dx + dy * dy + dz * dz;
      float dm = fminf(dist[i], d);
      dist[i] = dm;
      if (dm > best) { best = dm; bi = i; }
    }
#pragma unroll
    for (int m = 16; m > 0; m >>= 1) {
      float ov = __shfl_xor(best, m, 32);
      int   oi = __shfl_xor(bi, m, 32);
      if (ov > best || (ov == best && oi < bi)) { best = ov; bi = oi; }
    }
    if (lane == 0) { wmax[wid] = best; warg[wid] = bi; }
    __syncthreads();
    if (wid == 0) {
      float v = (lane < nw) ? wmax[lane] : -2.0f;
      int  vi = (lane < nw) ? warg[lane] : 0;
#pragma unroll
      for (int m = 16; m > 0; m >>= 1) {
        float ov = __shfl_xor(v, m, 32);
        int   oi = __shfl_xor(vi, m, 32);
        if (ov > v || (ov == v && oi < vi)) { v = ov; vi = oi; }
      }
      if (lane == 0) s_far = vi;
    }
    __syncthreads();
  }
}

__global__ void gather_xyz_kernel(const float* __restrict__ xyz, const int* __restrict__ fidx,
                                  int n, int npoint, float* __restrict__ out) {
  int t = blockIdx.x * blockDim.x + threadIdx.x;
  if (t >= BATCH * npoint) return;
  int b = t / npoint;
  int j = fidx[t];
  const float* p = xyz + ((size_t)b * n + j) * 3;
  float* o = out + (size_t)t * 3;
  o[0] = p[0]; o[1] = p[1]; o[2] = p[2];
}

// Ball query: first nsample ascending indices with d2 <= r2, padded with first.
__global__ void ball_query_kernel(const float* __restrict__ xyz, const float* __restrict__ newxyz,
                                  int n, int npoint, int nsample, float r2,
                                  int* __restrict__ gidx) {
  int t = blockIdx.x * blockDim.x + threadIdx.x;
  if (t >= BATCH * npoint) return;
  int b = t / npoint;
  const float* P = xyz + (size_t)b * n * 3;
  const float* C = newxyz + (size_t)t * 3;
  float cx = C[0], cy = C[1], cz = C[2];
  int* out = gidx + (size_t)t * nsample;
  int cnt = 0;
  for (int j = 0; j < n; ++j) {
    float dx = P[(size_t)j * 3] - cx;
    float dy = P[(size_t)j * 3 + 1] - cy;
    float dz = P[(size_t)j * 3 + 2] - cz;
    float d2 = dx * dx + dy * dy + dz * dz;
    if (d2 <= r2) {
      out[cnt++] = j;
      if (cnt == nsample) break;
    }
  }
  int f = (cnt > 0) ? out[0] : (n - 1);
  for (int k = cnt; k < nsample; ++k) out[k] = f;
}

// Build grouped f16 input (tiled): 3 recentered xyz channels + Cf features.
__global__ void build_group_in_kernel(const float* __restrict__ xyz, const float* __restrict__ newxyz,
                                      const int* __restrict__ gidx, const float* __restrict__ F,
                                      int Cf, int n, int npoint, int nsample, int Cpad,
                                      f16* __restrict__ Gin) {
  long long X = (long long)BATCH * npoint * nsample;
  long long x = (long long)blockIdx.x * blockDim.x + threadIdx.x;
  if (x >= X) return;
  long long t = x / nsample;
  int m = (int)(x % nsample);
  int s = (int)(t % npoint);
  int b = (int)(t / npoint);
  int j = gidx[((size_t)b * npoint + s) * nsample + m];
  const float* P = xyz + ((size_t)b * n + j) * 3;
  const float* C = newxyz + ((size_t)b * npoint + s) * 3;
  size_t cstride = (size_t)(X >> 4) * 512;
  f16* gx = Gin + (size_t)(x >> 4) * 512 + (size_t)(x & 15) * 32;
  auto put = [&](int c, float v) { gx[(size_t)(c >> 5) * cstride + (c & 31)] = (f16)v; };
  put(0, P[0] - C[0]);
  put(1, P[1] - C[1]);
  put(2, P[2] - C[2]);
  for (int c = 0; c < Cf; ++c)
    put(3 + c, F[(size_t)c * ((size_t)BATCH * n) + (size_t)b * n + j]);
  for (int c = 3 + Cf; c < Cpad; ++c) put(c, 0.0f);
}

// Max over nsample: tiled f16 act -> OutF f32 [Oreal][B*npoint]
__global__ void maxpool_kernel(const f16* __restrict__ Act, long long X, int nsample,
                               int npoint, int Oreal, float* __restrict__ OutF) {
  long long total = (long long)Oreal * BATCH * npoint;
  long long t = (long long)blockIdx.x * blockDim.x + threadIdx.x;
  if (t >= total) return;
  int s = (int)(t % npoint);
  long long u = t / npoint;
  int b = (int)(u % BATCH);
  int o = (int)(u / BATCH);
  size_t cb = (size_t)(o >> 5) * ((size_t)(X >> 4) * 512) + (size_t)(o & 31);
  long long xs = ((long long)b * npoint + s) * nsample;
  float mx = -1e30f;
  for (int k = 0; k < nsample; ++k) {
    long long x = xs + k;
    mx = fmaxf(mx, (float)Act[cb + (size_t)(x >> 4) * 512 + (size_t)(x & 15) * 32]);
  }
  OutF[(size_t)o * ((size_t)BATCH * npoint) + (size_t)b * npoint + s] = mx;
}

// 3-NN with inverse-distance weights
__global__ void knn3_kernel(const float* __restrict__ xyz1, const float* __restrict__ xyz2,
                            int n1, int n2, int* __restrict__ idx3, float* __restrict__ w3) {
  int t = blockIdx.x * blockDim.x + threadIdx.x;
  if (t >= BATCH * n1) return;
  int b = t / n1;
  const float* q = xyz1 + (size_t)t * 3;
  const float* P = xyz2 + (size_t)b * n2 * 3;
  float qx = q[0], qy = q[1], qz = q[2];
  float d0 = 1e30f, d1 = 1e30f, d2v = 1e30f;
  int i0 = 0, i1 = 0, i2 = 0;
  for (int j = 0; j < n2; ++j) {
    float dx = P[(size_t)j * 3] - qx;
    float dy = P[(size_t)j * 3 + 1] - qy;
    float dz = P[(size_t)j * 3 + 2] - qz;
    float d = dx * dx + dy * dy + dz * dz;
    if (d < d0)      { d2v = d1; i2 = i1; d1 = d0; i1 = i0; d0 = d; i0 = j; }
    else if (d < d1) { d2v = d1; i2 = i1; d1 = d;  i1 = j; }
    else if (d < d2v){ d2v = d;  i2 = j; }
  }
  float r0 = 1.0f / (d0 + 1e-8f);
  float r1 = 1.0f / (d1 + 1e-8f);
  float r2 = 1.0f / (d2v + 1e-8f);
  float s = r0 + r1 + r2;
  idx3[(size_t)t * 3 + 0] = i0; w3[(size_t)t * 3 + 0] = r0 / s;
  idx3[(size_t)t * 3 + 1] = i1; w3[(size_t)t * 3 + 1] = r1 / s;
  idx3[(size_t)t * 3 + 2] = i2; w3[(size_t)t * 3 + 2] = r2 / s;
}

// Build FP input (tiled): C1 channels from F1 + C2 interpolated from F2.
__global__ void build_fp_in_kernel(const float* __restrict__ F1, int C1,
                                   const float* __restrict__ F2, int C2,
                                   const int* __restrict__ idx3, const float* __restrict__ w3,
                                   int n1, int n2, int Cpad, f16* __restrict__ Out) {
  long long X = (long long)BATCH * n1;
  long long x = (long long)blockIdx.x * blockDim.x + threadIdx.x;
  if (x >= X) return;
  int b = (int)(x / n1);
  size_t cstride = (size_t)(X >> 4) * 512;
  f16* gx = Out + (size_t)(x >> 4) * 512 + (size_t)(x & 15) * 32;
  auto put = [&](int c, float v) { gx[(size_t)(c >> 5) * cstride + (c & 31)] = (f16)v; };
  for (int c = 0; c < C1; ++c)
    put(c, F1[(size_t)c * X + x]);
  int j0 = idx3[(size_t)x * 3 + 0], j1 = idx3[(size_t)x * 3 + 1], j2 = idx3[(size_t)x * 3 + 2];
  float w0 = w3[(size_t)x * 3 + 0], w1 = w3[(size_t)x * 3 + 1], w2 = w3[(size_t)x * 3 + 2];
  for (int c = 0; c < C2; ++c) {
    const float* basep = F2 + (size_t)c * ((size_t)BATCH * n2) + (size_t)b * n2;
    put(C1 + c, w0 * basep[j0] + w1 * basep[j1] + w2 * basep[j2]);
  }
  for (int c = C1 + C2; c < Cpad; ++c) put(c, 0.0f);
}

// log_softmax over channel dim; Lg [16][B*N] (rows >= NUM_CLASSES are zero padding)
__global__ void logsoftmax_kernel(const float* __restrict__ Lg, float* __restrict__ out) {
  int t = blockIdx.x * blockDim.x + threadIdx.x;
  if (t >= BATCH * NPTS) return;
  int b = t / NPTS, p = t % NPTS;
  const long long XT = (long long)BATCH * NPTS;
  float v[NUM_CLASSES];
  float mx = -1e30f;
  for (int o = 0; o < NUM_CLASSES; ++o) {
    v[o] = Lg[(size_t)o * XT + t];
    mx = fmaxf(mx, v[o]);
  }
  float s = 0.0f;
  for (int o = 0; o < NUM_CLASSES; ++o) s += __expf(v[o] - mx);
  float lse = mx + __logf(s);
  for (int o = 0; o < NUM_CLASSES; ++o)
    out[((size_t)b * NUM_CLASSES + o) * NPTS + p] = v[o] - lse;
}

// ---------------------------------------------------------------------------
// Host orchestration
// ---------------------------------------------------------------------------
struct LayerDef { const float *W, *b, *g, *be; int cin, cout; };

static inline int padC32(int c) { return (c + 31) & ~31; }
static inline int padO16(int c) { return (c + 15) & ~15; }
static inline int imax(int a, int b) { return a > b ? a : b; }

static void run_layer(const LayerDef& L, const f16* In, long long X,
                      f16* OutH, float* OutF, f16* Wp, hipStream_t stream) {
  int Opad = padO16(L.cout);
  int Cpad = padC32(L.cin);
  int tot = Opad * Cpad;
  convert_weight_kernel<<<(tot + 255) / 256, 256, 0, stream>>>(L.W, Wp, L.cout, L.cin, Opad, Cpad);
  int T = (Opad >= 64) ? 4 : ((Opad >= 32) ? 2 : 1);
  dim3 blk(32, 4);
  dim3 grd((unsigned)(X / 64), (unsigned)(Opad / (16 * T)));
  if (T == 4)
    gemm_bn_relu_kernel<4><<<grd, blk, 0, stream>>>(Wp, In, L.b, L.g, L.be, OutH, OutF,
                                                    Opad, Cpad, X, L.cout);
  else if (T == 2)
    gemm_bn_relu_kernel<2><<<grd, blk, 0, stream>>>(Wp, In, L.b, L.g, L.be, OutH, OutF,
                                                    Opad, Cpad, X, L.cout);
  else
    gemm_bn_relu_kernel<1><<<grd, blk, 0, stream>>>(Wp, In, L.b, L.g, L.be, OutH, OutF,
                                                    Opad, Cpad, X, L.cout);
}

extern "C" void kernel_launch(void* const* d_in, const int* in_sizes, int n_in,
                              void* d_out, int out_size, void* d_ws, size_t ws_size,
                              hipStream_t stream) {
  (void)in_sizes; (void)n_in; (void)out_size; (void)ws_size;
  const float* x = (const float*)d_in[0];

  // --- parameter mapping (pytree flatten, dicts sorted by key) ---
  int cur = 1;
  auto readLayer = [&](int cin, int cout) -> LayerDef {
    LayerDef l; l.cin = cin; l.cout = cout;
    l.W  = (const float*)d_in[cur++];
    l.b  = (const float*)d_in[cur++];
    l.be = (const float*)d_in[cur++];   // 'beta'
    l.g  = (const float*)d_in[cur++];   // 'g'
    return l;
  };
  LayerDef fp1[3] = { readLayer(128,128), readLayer(128,128), readLayer(128,128) };
  LayerDef fp2[2] = { readLayer(320,256), readLayer(256,128) };
  LayerDef fp3[2] = { readLayer(384,256), readLayer(256,256) };
  LayerDef h1 = readLayer(128,128);
  LayerDef h2; h2.cin = 128; h2.cout = NUM_CLASSES;
  h2.W = (const float*)d_in[cur++]; h2.b = (const float*)d_in[cur++];
  h2.g = nullptr; h2.be = nullptr;
  LayerDef sa1[3] = { readLayer(3,32),   readLayer(32,32),  readLayer(32,64)  };
  LayerDef sa2[3] = { readLayer(67,64),  readLayer(64,64),  readLayer(64,128) };
  LayerDef sa3[3] = { readLayer(131,128),readLayer(128,128),readLayer(128,256)};

  // --- workspace bump allocator ---
  char* base = (char*)d_ws;
  size_t off = 0;
  auto alloc = [&](size_t bytes) -> void* {
    void* p = base + off;
    off += (bytes + 255) & ~(size_t)255;
    return p;
  };
  // persistent
  float* xyz0  = (float*)alloc((size_t)BATCH * NPTS * 3 * 4);
  float* l1xyz = (float*)alloc((size_t)BATCH * 1024 * 3 * 4);
  float* l2xyz = (float*)alloc((size_t)BATCH * 256 * 3 * 4);
  float* l3xyz = (float*)alloc((size_t)BATCH * 64 * 3 * 4);
  int* fidx1 = (int*)alloc((size_t)BATCH * 1024 * 4);
  int* fidx2 = (int*)alloc((size_t)BATCH * 256 * 4);
  int* fidx3 = (int*)alloc((size_t)BATCH * 64 * 4);
  float* f1  = (float*)alloc((size_t)64  * BATCH * 1024 * 4);  // sa1 out
  float* f2  = (float*)alloc((size_t)128 * BATCH * 256  * 4);  // sa2 out
  float* f3  = (float*)alloc((size_t)256 * BATCH * 64   * 4);  // sa3 out
  float* f2n = (float*)alloc((size_t)256 * BATCH * 256  * 4);  // fp3 out
  float* f1n = (float*)alloc((size_t)128 * BATCH * 1024 * 4);  // fp2 out
  f16* Wp = (f16*)alloc((size_t)256 * 384 * 2);                // weight scratch
  const size_t arena0 = off;

  // --- set abstraction stage ---
  auto run_sa = [&](const float* xyz, int n, const float* feat, int Cf,
                    int npoint, int nsample, float radius,
                    const LayerDef* Ls, int nl,
                    int* fidx, float* outxyz, float* outfeat) {
    off = arena0;
    long long X = (long long)BATCH * npoint * nsample;
    int Cp = padC32(3 + Cf);
    int maxC = Cp;
    for (int i = 0; i < nl; ++i) maxC = imax(maxC, padO16(Ls[i].cout));
    f16* actA = (f16*)alloc((size_t)maxC * X * 2);
    f16* actB = (f16*)alloc((size_t)maxC * X * 2);
    int* gidx = (int*)alloc((size_t)BATCH * npoint * nsample * 4);

    int thr = (n >= 4096) ? 1024 : 256;
    size_t sh = (size_t)n * 4 + 32 * 8;
    fps_kernel<<<BATCH, thr, sh, stream>>>(xyz, n, npoint, fidx);
    gather_xyz_kernel<<<(BATCH * npoint + 255) / 256, 256, 0, stream>>>(xyz, fidx, n, npoint, outxyz);
    ball_query_kernel<<<(BATCH * npoint + 255) / 256, 256, 0, stream>>>(
        xyz, outxyz, n, npoint, nsample, radius * radius, gidx);
    build_group_in_kernel<<<(unsigned)((X + 255) / 256), 256, 0, stream>>>(
        xyz, outxyz, gidx, feat, Cf, n, npoint, nsample, Cp, actA);

    const f16* in = actA; f16* out = actB;
    for (int i = 0; i < nl; ++i) {
      run_layer(Ls[i], in, X, out, nullptr, Wp, stream);
      f16* t = (f16*)in; in = out; out = t;
    }
    int Oreal = Ls[nl - 1].cout;
    long long tot = (long long)Oreal * BATCH * npoint;
    maxpool_kernel<<<(unsigned)((tot + 255) / 256), 256, 0, stream>>>(
        in, X, nsample, npoint, Oreal, outfeat);
  };

  // --- feature propagation stage; returns final f16 act + spare buffer ---
  struct FPRes { f16* last; f16* other; };
  auto run_fp = [&](const float* xyz1, int n1, const float* xyz2, int n2,
                    const float* F1, int C1, const float* F2, int C2,
                    const LayerDef* Ls, int nl, float* outF) -> FPRes {
    off = arena0;
    long long X = (long long)BATCH * n1;
    int Cp = padC32(C1 + C2);
    int maxC = Cp;
    for (int i = 0; i < nl; ++i) maxC = imax(maxC, padO16(Ls[i].cout));
    f16* actA = (f16*)alloc((size_t)maxC * X * 2);
    f16* actB = (f16*)alloc((size_t)maxC * X * 2);
    int* idx3  = (int*)alloc((size_t)BATCH * n1 * 3 * 4);
    float* w3  = (float*)alloc((size_t)BATCH * n1 * 3 * 4);

    knn3_kernel<<<(BATCH * n1 + 255) / 256, 256, 0, stream>>>(xyz1, xyz2, n1, n2, idx3, w3);
    build_fp_in_kernel<<<(unsigned)((X + 255) / 256), 256, 0, stream>>>(
        F1, C1, F2, C2, idx3, w3, n1, n2, Cp, actA);

    const f16* in = actA; f16* out = actB;
    for (int i = 0; i < nl; ++i) {
      float* of = (i == nl - 1) ? outF : nullptr;
      run_layer(Ls[i], in, X, out, of, Wp, stream);
      f16* t = (f16*)in; in = out; out = t;
    }
    FPRes r; r.last = (f16*)in; r.other = out;
    return r;
  };

  // ---------------- pipeline ----------------
  transpose_xyz_kernel<<<(BATCH * NPTS + 255) / 256, 256, 0, stream>>>(x, xyz0);

  run_sa(xyz0,  NPTS, nullptr, 0,   1024, 32,  1.0f, sa1, 3, fidx1, l1xyz, f1);
  run_sa(l1xyz, 1024, f1,      64,  256,  64,  3.0f, sa2, 3, fidx2, l2xyz, f2);
  run_sa(l2xyz, 256,  f2,      128, 64,   256, 9.0f, sa3, 3, fidx3, l3xyz, f3);

  run_fp(l2xyz, 256,  l3xyz, 64,  f2, 128, f3,  256, fp3, 2, f2n);
  run_fp(l1xyz, 1024, l2xyz, 256, f1, 64,  f2n, 256, fp2, 2, f1n);
  FPRes r = run_fp(xyz0, NPTS, l1xyz, 1024, nullptr, 0, f1n, 128, fp1, 3, nullptr);

  // head: h1 (relu) then h2 (logits) then log_softmax
  long long X0 = (long long)BATCH * NPTS;
  run_layer(h1, r.last, X0, r.other, nullptr, Wp, stream);
  float* logits = (float*)alloc((size_t)16 * X0 * 4);
  run_layer(h2, r.other, X0, nullptr, logits, Wp, stream);
  logsoftmax_kernel<<<(unsigned)((X0 + 255) / 256), 256, 0, stream>>>(logits, (float*)d_out);
}
